// SAGEConv_Encoder_22316650070979
// MI455X (gfx1250) — matrix operand order repaired and checked by reading, not compiled, not run on hardware
//
#include <hip/hip_runtime.h>
#include <hip/hip_bf16.h>
#include <math.h>

typedef _Float16 v16h __attribute__((ext_vector_type(16)));
typedef _Float16 v8h  __attribute__((ext_vector_type(8)));
typedef float    v8f  __attribute__((ext_vector_type(8)));

#define C 128   // IN_C == OUT_C == 128

// ---------------------------------------------------------------- utilities

__global__ void cvt_w_kernel(const float* __restrict__ w,
                             _Float16* __restrict__ w16, int n) {
    int i = blockIdx.x * blockDim.x + threadIdx.x;
    if (i < n) w16[i] = (_Float16)w[i];
}

__global__ void deg_kernel(const int* __restrict__ dst,
                           float* __restrict__ deg, int E) {
    int e = blockIdx.x * blockDim.x + threadIdx.x;
    if (e < E) unsafeAtomicAdd(&deg[dst[e]], 1.0f);
}

__global__ void invdeg_kernel(float* __restrict__ deg, int N) {
    int i = blockIdx.x * blockDim.x + threadIdx.x;
    if (i < N) deg[i] = 1.0f / fmaxf(deg[i], 1.0f);
}

// One wave per edge: 32 lanes x float4 = 128 channels gathered from x[src],
// scatter-added into agg[dst] with native global_atomic_add_f32 (L2-resident).
__global__ void scatter_kernel(const float* __restrict__ X,
                               const int* __restrict__ src,
                               const int* __restrict__ dst,
                               float* __restrict__ agg, int E) {
    int lane = threadIdx.x & 31;
    int e = (int)((blockIdx.x * (unsigned)blockDim.x + threadIdx.x) >> 5);
    if (e >= E) return;
    int s = src[e];
    int d = dst[e];
    float4 v = ((const float4*)(X + (size_t)s * C))[lane];
    float* out = agg + (size_t)d * C + lane * 4;
    unsafeAtomicAdd(out + 0, v.x);
    unsafeAtomicAdd(out + 1, v.y);
    unsafeAtomicAdd(out + 2, v.z);
    unsafeAtomicAdd(out + 3, v.w);
}

// ---------------------------------------------------------------- WMMA GEMM

// A-fragment (16x32 f16): lane holds row m=lane%16,
// k = 16*half + j (j<16), matching B: two contiguous 16B reads.
__device__ inline v16h frag_a(const _Float16* __restrict__ row, int half) {
    v8h lo = *(const v8h*)(row + half * 16);
    v8h hi = *(const v8h*)(row + half * 16 + 8);
    v16h a;
#pragma unroll
    for (int j = 0; j < 8; ++j) { a[j] = lo[j]; a[j + 8] = hi[j]; }
    return a;
}

// out[m,n] = sum_k (agg[m,k]*invdeg[m]) * Wl[n,k] + bl[n] + sum_k X[m,k]*Wr[n,k]
// then row-wise L2 normalize.  Block = 256 threads = 8 waves; block covers a
// 16-node tile, wave w covers out channels [16w,16w+16), K=128 in 4 WMMA steps
// per matrix product (8 v_wmma_f32_16x16x32_f16 per wave total).
__global__ __launch_bounds__(256) void sage_gemm_kernel(
        const float* __restrict__ agg, const float* __restrict__ invdeg,
        const float* __restrict__ X,
        const _Float16* __restrict__ Wl, const float* __restrict__ bl,
        const _Float16* __restrict__ Wr,
        float* __restrict__ Out, int N) {
    __shared__ _Float16 lA[16 * C];   // mean-aggregated features, f16
    __shared__ _Float16 lX[16 * C];   // self features, f16
    __shared__ float    sums[16];     // per-row sum of squares
    __shared__ float    stage[16 * C];

    const int tid  = threadIdx.x;
    const int base = blockIdx.x * 16;

    if (tid < 16) sums[tid] = 0.0f;
#pragma unroll
    for (int j = 0; j < 8; ++j) {
        int i = tid + 256 * j;            // 2048 elements, coalesced
        int row = i >> 7;
        int k = i & (C - 1);
        int r = base + row; if (r >= N) r = N - 1;
        lA[i] = (_Float16)(agg[(size_t)r * C + k] * invdeg[r]);
        lX[i] = (_Float16)(X[(size_t)r * C + k]);
    }
    __syncthreads();

    const int lane = tid & 31;
    const int w    = tid >> 5;        // wave id: which 16-wide N tile
    const int half = lane >> 4;
    const int nl   = lane & 15;
    const int n    = w * 16 + nl;     // output channel for this lane

    const _Float16* wl_row = Wl + (size_t)n * C;   // B: Wl[n,k], k contiguous
    const _Float16* wr_row = Wr + (size_t)n * C;
    const _Float16* arow   = lA + nl * C;          // A row m = lane%16
    const _Float16* xrow   = lX + nl * C;

    v8f c = {};
#pragma unroll
    for (int t = 0; t < 4; ++t) {
        int kb = t * 32;
        v16h a = frag_a(arow + kb, half);
        v16h b = *(const v16h*)(wl_row + kb + 16 * half);  // B 32x16: k=16*half+j
        c = __builtin_amdgcn_wmma_f32_16x16x32_f16(false, a, false, b,
                                                   (short)0, c, false, false);
    }
#pragma unroll
    for (int t = 0; t < 4; ++t) {
        int kb = t * 32;
        v16h a = frag_a(xrow + kb, half);
        v16h b = *(const v16h*)(wr_row + kb + 16 * half);
        c = __builtin_amdgcn_wmma_f32_16x16x32_f16(false, a, false, b,
                                                   (short)0, c, false, false);
    }

    // bias + squared-sum reduction.  D layout: VGPR r -> M = r + 8*half,
    // N = lane%16; reduce over the 16 lanes of each half, then over waves via
    // LDS float atomics.
    const float bias = bl[n];
#pragma unroll
    for (int r = 0; r < 8; ++r) {
        c[r] += bias;
        float v = c[r] * c[r];
        v += __shfl_xor(v, 1, 32);
        v += __shfl_xor(v, 2, 32);
        v += __shfl_xor(v, 4, 32);
        v += __shfl_xor(v, 8, 32);
        if (nl == 0) atomicAdd(&sums[r + 8 * half], v);
    }
    __syncthreads();

#pragma unroll
    for (int r = 0; r < 8; ++r) {
        int m = r + 8 * half;
        float inv = 1.0f / fmaxf(sqrtf(sums[m]), 1e-12f);
        stage[m * C + n] = c[r] * inv;
    }
    __syncthreads();

#pragma unroll
    for (int j = 0; j < 8; ++j) {
        int i = tid + 256 * j;
        int row = i >> 7;
        if (base + row < N) Out[(size_t)base * C + i] = stage[i];
    }
}

// ---------------------------------------------------------------- launcher

extern "C" void kernel_launch(void* const* d_in, const int* in_sizes, int n_in,
                              void* d_out, int out_size, void* d_ws, size_t ws_size,
                              hipStream_t stream) {
    const float* x   = (const float*)d_in[0];
    const int*   ei  = (const int*)d_in[1];
    const float* Wl1 = (const float*)d_in[2];
    const float* bl1 = (const float*)d_in[3];
    const float* Wr1 = (const float*)d_in[4];
    const float* Wl2 = (const float*)d_in[5];
    const float* bl2 = (const float*)d_in[6];
    const float* Wr2 = (const float*)d_in[7];

    const int N = in_sizes[0] / C;
    const int E = in_sizes[1] / 2;
    const int* src = ei;
    const int* dst = ei + E;

    // workspace layout
    char* ws = (char*)d_ws;
    const size_t aggB = (size_t)N * C * sizeof(float);
    const size_t degB = (((size_t)N * sizeof(float)) + 255) & ~(size_t)255;
    float*    agg = (float*)ws;
    float*    h   = (float*)(ws + aggB);
    float*    deg = (float*)(ws + 2 * aggB);
    _Float16* w16 = (_Float16*)(ws + 2 * aggB + degB);
    _Float16* wl1_16 = w16;
    _Float16* wr1_16 = w16 + 1 * C * C;
    _Float16* wl2_16 = w16 + 2 * C * C;
    _Float16* wr2_16 = w16 + 3 * C * C;

    hipMemsetAsync(deg, 0, (size_t)N * sizeof(float), stream);
    hipMemsetAsync(agg, 0, aggB, stream);

    const int WN = C * C;
    cvt_w_kernel<<<(WN + 255) / 256, 256, 0, stream>>>(Wl1, wl1_16, WN);
    cvt_w_kernel<<<(WN + 255) / 256, 256, 0, stream>>>(Wr1, wr1_16, WN);
    cvt_w_kernel<<<(WN + 255) / 256, 256, 0, stream>>>(Wl2, wl2_16, WN);
    cvt_w_kernel<<<(WN + 255) / 256, 256, 0, stream>>>(Wr2, wr2_16, WN);

    deg_kernel<<<(E + 255) / 256, 256, 0, stream>>>(dst, deg, E);
    invdeg_kernel<<<(N + 255) / 256, 256, 0, stream>>>(deg, N);

    const int scatterBlocks = (E + 7) / 8;   // 8 edges (waves) per 256-thread block

    // ---- layer 1
    scatter_kernel<<<scatterBlocks, 256, 0, stream>>>(x, src, dst, agg, E);
    sage_gemm_kernel<<<(N + 15) / 16, 256, 0, stream>>>(
        agg, deg, x, wl1_16, bl1, wr1_16, h, N);

    // ---- layer 2
    hipMemsetAsync(agg, 0, aggB, stream);
    scatter_kernel<<<scatterBlocks, 256, 0, stream>>>(h, src, dst, agg, E);
    sage_gemm_kernel<<<(N + 15) / 16, 256, 0, stream>>>(
        agg, deg, h, wl2_16, bl2, wr2_16, (float*)d_out, N);
}
